// SeqDecoder_12412455486034
// MI455X (gfx1250) — matrix operand order repaired and checked by reading, not compiled, not run on hardware
//
#include <hip/hip_runtime.h>
#include <hip/hip_bf16.h>

// SeqDecoder (LSTM decode loop) for MI455X / gfx1250.
// B=64, H=512, V=32000, T=24 (matches reference setup_inputs()).
//
// Key insight: x is a hard one-hot after step 0, so x @ W_ih.T is a column
// gather of W_ih -- NOT a dense [B,V]x[V,4H] GEMM. The only real GEMMs are
// h @ W_hh.T ([64x512]x[512x2048]) and h_new @ W_out.T ([64x512]x[512x32000]),
// run as wave-per-16x16-tile v_wmma_f32_16x16x32_f16 with f32 accumulation.
// W_out in f16 (32.8MB) stays resident in the 192MB L2 across all 24 steps.

typedef __attribute__((ext_vector_type(16))) _Float16 v16h;
typedef __attribute__((ext_vector_type(8)))  _Float16 v8h;
typedef __attribute__((ext_vector_type(8)))  float    v8f;

#define B_  64
#define H_  512
#define V_  32000
#define T_  24
#define G_  (4 * H_)      // 2048
#define EOS_IDX_ 0

__device__ __forceinline__ float sigmoidf_(float x) {
  return 1.0f / (1.0f + __expf(-x));
}

// ---------------- utility kernels ----------------

__global__ void k_f32_to_f16(const float* __restrict__ src,
                             _Float16* __restrict__ dst, int n) {
  int i = blockIdx.x * blockDim.x + threadIdx.x;
  int stride = gridDim.x * blockDim.x;
  for (; i < n; i += stride) dst[i] = (_Float16)src[i];
}

__global__ void k_zero_f4(float4* __restrict__ p, int n4) {
  int i = blockIdx.x * blockDim.x + threadIdx.x;
  int stride = gridDim.x * blockDim.x;
  float4 z; z.x = 0.f; z.y = 0.f; z.z = 0.f; z.w = 0.f;
  for (; i < n4; i += stride) p[i] = z;
}

__global__ void k_init_state(const float* __restrict__ eh,
                             const float* __restrict__ ec,
                             float* __restrict__ c,
                             _Float16* __restrict__ hf16,
                             float* __restrict__ mask_state,
                             int* __restrict__ idx) {
  int i = blockIdx.x * blockDim.x + threadIdx.x;
  if (i < B_ * H_) {
    c[i]    = ec[i];
    hf16[i] = (_Float16)eh[i];
  }
  if (i < B_) {
    mask_state[i] = 1.0f;
    idx[i]        = -1;   // step 0: x is all-zeros -> no W_ih gather
  }
}

// ---------------- WMMA tile engine ----------------
// Computes one 16x16 f32 tile of  A[64xK](f16, row-major, ld=lda)
//                               x W[NxK](f16, row-major, ld=ldw)^T
// i.e. D[m,n] = sum_k A[m0+m, k] * W[n0+n, k].
//
// Per-lane operand packing follows the CDNA5 ISA VGPR layout tables:
//   A (16x32 f16): lanes 0-15 -> K {0..7, 16..23}; lanes 16-31 -> K {8..15, 24..31}
//   B (32x16 f16): n = lane&15; K order same as A: lanes 0-15 -> K {0..7, 16..23}; lanes 16-31 -> K {8..15, 24..31}
//   D (16x16 f32): m = vgpr + 8*(lane>>4); n = lane&15

__device__ __forceinline__ v8f wmma_tile_64xK(const _Float16* __restrict__ A, int lda,
                                              const _Float16* __restrict__ W, int ldw,
                                              int m0, int n0, int K) {
  const int lane = threadIdx.x & 31;
  const int hsel = lane >> 4;   // half-wave select
  const int lrow = lane & 15;
  const _Float16* aP = A + (size_t)(m0 + lrow) * lda;
  const _Float16* bP = W + (size_t)(n0 + lrow) * ldw;
  v8f acc = {};
#pragma unroll 8
  for (int kb = 0; kb < K; kb += 32) {
    v8h a0 = *(const v8h*)(aP + kb + hsel * 8);        // 16B global_load_b128
    v8h a1 = *(const v8h*)(aP + kb + 16 + hsel * 8);
    v8h b0 = *(const v8h*)(bP + kb + hsel * 8);
    v8h b1 = *(const v8h*)(bP + kb + 16 + hsel * 8);
    v16h a, b;
#pragma unroll
    for (int j = 0; j < 8; ++j) {
      a[j] = a0[j]; a[j + 8] = a1[j];
      b[j] = b0[j]; b[j + 8] = b1[j];
    }
    acc = __builtin_amdgcn_wmma_f32_16x16x32_f16(
        /*neg_a=*/false, a, /*neg_b=*/false, b,
        /*c_mod=*/(short)0, acc, /*reuse_a=*/false, /*reuse_b=*/false);
  }
  return acc;
}

// ---------------- gates GEMM: gates = h @ W_hh^T + (b_ih + b_hh) + W_ih[:, idx] ----------------
// grid: 64 blocks x 256 thr = 512 waves = 4 (M tiles) x 128 (N tiles)

__global__ void k_gates(const _Float16* __restrict__ hf16,
                        const _Float16* __restrict__ whh16,
                        const float* __restrict__ Wih,   // [4H, V] f32, gathered only
                        const float* __restrict__ bih,
                        const float* __restrict__ bhh,
                        const int* __restrict__ idx,
                        float* __restrict__ gates) {     // [B, 4H]
  const int wave   = (blockIdx.x * blockDim.x + threadIdx.x) >> 5;
  const int tile_m = wave & 3;
  const int tile_n = wave >> 2;
  const int m0 = tile_m * 16, n0 = tile_n * 16;
  v8f acc = wmma_tile_64xK(hf16, H_, whh16, H_, m0, n0, H_);
  const int lane = threadIdx.x & 31;
  const int hsel = lane >> 4, lrow = lane & 15;
  const int n = n0 + lrow;
  const float bn = bih[n] + bhh[n];
#pragma unroll
  for (int r = 0; r < 8; ++r) {
    const int m = m0 + r + hsel * 8;
    float g = acc[r] + bn;
    const int id = idx[m];
    if (id >= 0) g += Wih[(size_t)n * V_ + id];   // one-hot x contribution
    gates[m * G_ + n] = g;
  }
}

// ---------------- LSTM cell pointwise ----------------

__global__ void k_cell(const float* __restrict__ gates,
                       float* __restrict__ c,
                       _Float16* __restrict__ hf16) {
  int i = blockIdx.x * blockDim.x + threadIdx.x;
  if (i >= B_ * H_) return;
  const int b = i >> 9;           // / H_
  const int j = i & (H_ - 1);
  const float* g = gates + b * G_;
  const float gi = g[j], gf = g[H_ + j], gg = g[2 * H_ + j], go = g[3 * H_ + j];
  const float cn = sigmoidf_(gf) * c[i] + sigmoidf_(gi) * tanhf(gg);
  const float hn = sigmoidf_(go) * tanhf(cn);
  c[i]    = cn;
  hf16[i] = (_Float16)hn;
}

// ---------------- logits GEMM: logits = h_new @ W_out^T + b_out ----------------
// grid: 1000 blocks x 256 thr = 8000 waves = 4 (M tiles) x 2000 (N tiles)

__global__ void k_logits(const _Float16* __restrict__ hf16,
                         const _Float16* __restrict__ wout16,
                         const float* __restrict__ bout,
                         float* __restrict__ out) {       // [B, V] slice of d_out
  const int wave   = (blockIdx.x * blockDim.x + threadIdx.x) >> 5;
  const int tile_m = wave & 3;
  const int tile_n = wave >> 2;
  const int m0 = tile_m * 16, n0 = tile_n * 16;
  v8f acc = wmma_tile_64xK(hf16, H_, wout16, H_, m0, n0, H_);
  const int lane = threadIdx.x & 31;
  const int hsel = lane >> 4, lrow = lane & 15;
  const int n = n0 + lrow;
  const float bn = bout[n];
#pragma unroll
  for (int r = 0; r < 8; ++r) {
    const int m = m0 + r + hsel * 8;
    out[(size_t)m * V_ + n] = acc[r] + bn;
  }
}

// ---------------- argmax + one-hot + mask + next-step index ----------------
// one block per batch row

__global__ void k_argmax(const float* __restrict__ logit_t,  // [B, V]
                         float* __restrict__ pred_t,         // [B, V] (pre-zeroed)
                         float* __restrict__ mask_t,         // [B] -> masks[t,0,:]
                         float* __restrict__ mask_state,     // [B] running state
                         int* __restrict__ idx) {            // [B] next gather idx
  __shared__ float sv[256];
  __shared__ int   si[256];
  const int b = blockIdx.x;
  const float* row = logit_t + (size_t)b * V_;
  float best = -3.402823466e38f;
  int   bi   = 0;
  for (int j = threadIdx.x; j < V_; j += 256) {
    const float v = row[j];
    if (v > best) { best = v; bi = j; }
  }
  sv[threadIdx.x] = best; si[threadIdx.x] = bi;
  __syncthreads();
  for (int s = 128; s > 0; s >>= 1) {
    if (threadIdx.x < s) {
      const float v2 = sv[threadIdx.x + s];
      const int   i2 = si[threadIdx.x + s];
      if (v2 > sv[threadIdx.x] ||
          (v2 == sv[threadIdx.x] && i2 < si[threadIdx.x])) {   // first-index ties
        sv[threadIdx.x] = v2; si[threadIdx.x] = i2;
      }
    }
    __syncthreads();
  }
  if (threadIdx.x == 0) {
    const int am = si[0];
    mask_t[b] = mask_state[b];                 // reference emits the INCOMING mask
    pred_t[(size_t)b * V_ + am] = 1.0f;
    if (am == EOS_IDX_) mask_state[b] = 0.0f;  // mask *= (1 - predict[:,EOS])
    idx[b] = am;
  }
}

// ---------------- host launcher ----------------

extern "C" void kernel_launch(void* const* d_in, const int* in_sizes, int n_in,
                              void* d_out, int out_size, void* d_ws, size_t ws_size,
                              hipStream_t stream) {
  const float* eh   = (const float*)d_in[0];  // encoder_hidden [B,H]
  const float* ec   = (const float*)d_in[1];  // encoder_cell   [B,H]
  const float* Wih  = (const float*)d_in[2];  // [4H, V]
  const float* Whh  = (const float*)d_in[3];  // [4H, H]
  const float* bih  = (const float*)d_in[4];  // [4H]
  const float* bhh  = (const float*)d_in[5];  // [4H]
  const float* Wout = (const float*)d_in[6];  // [V, H]
  const float* bout = (const float*)d_in[7];  // [V]
  (void)in_sizes; (void)n_in; (void)out_size;

  float* out      = (float*)d_out;
  float* predicts = out;                                  // [T,B,V]
  float* logits   = out + (size_t)T_ * B_ * V_;           // [T,B,V]
  float* masks    = out + 2 * (size_t)T_ * B_ * V_;       // [T,1,B]

  // workspace carve (total ~35.7 MB)
  char* ws = (char*)d_ws;
  auto carve = [&](size_t bytes) {
    char* p = ws;
    ws += (bytes + 255) & ~(size_t)255;
    return p;
  };
  _Float16* whh16      = (_Float16*)carve((size_t)G_ * H_ * sizeof(_Float16));
  _Float16* wout16     = (_Float16*)carve((size_t)V_ * H_ * sizeof(_Float16));
  float*    gates      = (float*)   carve((size_t)B_ * G_ * sizeof(float));
  _Float16* hf16       = (_Float16*)carve((size_t)B_ * H_ * sizeof(_Float16));
  float*    c          = (float*)   carve((size_t)B_ * H_ * sizeof(float));
  float*    mask_state = (float*)   carve(B_ * sizeof(float));
  int*      idx        = (int*)     carve(B_ * sizeof(int));
  (void)ws_size;

  // one-time (per launch) prep: f16 weight copies, zero one-hot region, init state
  k_f32_to_f16<<<256,  256, 0, stream>>>(Whh,  whh16,  G_ * H_);
  k_f32_to_f16<<<1024, 256, 0, stream>>>(Wout, wout16, V_ * H_);
  k_zero_f4  <<<2048,  256, 0, stream>>>((float4*)predicts, (T_ * B_ * V_) / 4);
  k_init_state<<<(B_ * H_ + 255) / 256, 256, 0, stream>>>(eh, ec, c, hf16, mask_state, idx);

  const int gatesBlocks  = (4 * (G_ / 16)) / 8;   // 64 blocks  (512 wave-tiles)
  const int logitsBlocks = (4 * (V_ / 16)) / 8;   // 1000 blocks (8000 wave-tiles)

  for (int t = 0; t < T_; ++t) {
    float* logit_t = logits   + (size_t)t * B_ * V_;
    float* pred_t  = predicts + (size_t)t * B_ * V_;
    float* mask_t  = masks    + (size_t)t * B_;
    k_gates <<<gatesBlocks,  256, 0, stream>>>(hf16, whh16, Wih, bih, bhh, idx, gates);
    k_cell  <<<(B_ * H_ + 255) / 256, 256, 0, stream>>>(gates, c, hf16);
    k_logits<<<logitsBlocks, 256, 0, stream>>>(hf16, wout16, bout, logit_t);
    k_argmax<<<B_, 256, 0, stream>>>(logit_t, pred_t, mask_t, mask_state, idx);
  }
}
